// HeavilyCompressedAttention_70643622085011
// MI455X (gfx1250) — matrix operand order repaired and checked by reading, not compiled, not run on hardware
//
#include <hip/hip_runtime.h>
#include <hip/hip_bf16.h>

// ---------------------------------------------------------------------------
// Types
// ---------------------------------------------------------------------------
typedef __bf16 bf16;
typedef __attribute__((ext_vector_type(16))) __bf16 v16bf;
typedef __attribute__((ext_vector_type(8)))  __bf16 v8bf;
typedef __attribute__((ext_vector_type(8)))  float  v8f;
typedef unsigned int u32x4_t __attribute__((ext_vector_type(4)));
typedef int          i32x8_t __attribute__((ext_vector_type(8)));
typedef int          i32x4_t __attribute__((ext_vector_type(4)));

#define B_SZ     2
#define S_SZ     8192
#define D_MODEL  2048
#define M_BLK    128
#define N_H      16
#define C_DIM    128
#define D_C      512
#define N_GROUPS 4
#define D_G      512
#define N_COMP   (S_SZ / M_BLK)       // 64
#define ROWS     (B_SZ * S_SZ)        // 16384

// ---------------------------------------------------------------------------
// WMMA fragment helpers (CDNA5 wave32; A/B k order is this file's own choice, not the ISA table order)
//   A 16x32 bf16 : lane (r=lane&15, h=lane>>4) holds row r,
//                  K = {kb+16h .. kb+16h+15} (contiguous, matches B)
//   B 32x16 bf16 : lane holds col r, K = {kb+16h .. kb+16h+15} (contiguous)
//                  -> we stage B transposed [N][K] so this is one 32B read
//   C 16x16 f32  : acc[i] -> row (i + 8h), col r
// ---------------------------------------------------------------------------
__device__ __forceinline__ v16bf cat8(v8bf lo, v8bf hi) {
    return __builtin_shufflevector(lo, hi, 0, 1, 2, 3, 4, 5, 6, 7,
                                   8, 9, 10, 11, 12, 13, 14, 15);
}

__device__ __forceinline__ v16bf frag_a_ptr(const bf16* rowp, int kb) {
    const int hl = (threadIdx.x >> 4) & 1;
    return cat8(*(const v8bf*)(rowp + kb + 16 * hl),
                *(const v8bf*)(rowp + kb + 16 * hl + 8));
}

__device__ __forceinline__ v16bf frag_b_ptr(const bf16* base, int stride, int kb) {
    const int lane = threadIdx.x & 31;
    const bf16* p = base + (lane & 15) * stride + kb + 16 * (lane >> 4);
    return cat8(*(const v8bf*)(p), *(const v8bf*)(p + 8));
}

__device__ __forceinline__ v8f wmma_bf16(v16bf a, v16bf b, v8f c) {
    return __builtin_amdgcn_wmma_f32_16x16x32_bf16(false, a, false, b, (short)0, c,
                                                   false, false);
}

// ---------------------------------------------------------------------------
// f32 -> bf16 cast kernels
// ---------------------------------------------------------------------------
__global__ void __launch_bounds__(256)
cast_f32_bf16_kernel(const float* __restrict__ s, bf16* __restrict__ d, long n) {
    long i = (long)blockIdx.x * 256 + threadIdx.x;
    if (i < n) d[i] = (bf16)s[i];
}

// wcat[k][0:128] = w_kv[k], wcat[k][128:256] = w_z[k]
__global__ void __launch_bounds__(256)
build_wcat_kernel(const float* __restrict__ wkv, const float* __restrict__ wz,
                  bf16* __restrict__ dst) {
    long i = (long)blockIdx.x * 256 + threadIdx.x;
    if (i >= (long)D_MODEL * 256) return;
    int row = (int)(i >> 8), col = (int)(i & 255);
    float v = (col < 128) ? wkv[row * 128 + col] : wz[row * 128 + (col - 128)];
    dst[i] = (bf16)v;
}

// ---------------------------------------------------------------------------
// Generic bf16 WMMA GEMM: C[M x N] = A[M x K] * B[K x N]  (row-major)
// block = 256 threads (8 waves), 64x128 tile, waves 2x4, each wave owns a
// 32x32 output (2 A-frags x 2 B-frags = 4 WMMAs / K-step of 32).
// Register double-buffering: global loads for step k+1 issue before the
// WMMAs of step k consume LDS, so s_wait_loadcnt lands after compute.
// ---------------------------------------------------------------------------
__global__ void __launch_bounds__(256)
gemm_bf16_kernel(const bf16* __restrict__ A, const bf16* __restrict__ B,
                 float* __restrict__ Cf, bf16* __restrict__ Cb,
                 int K, int lda, int ldb, int ldc,
                 long sA, long sB, long sC) {
    __shared__ bf16 As[64][40];    // 64x32 A tile, padded rows (80B)
    __shared__ bf16 Bt[128][40];   // 32x128 B tile, transposed [N][K]

    const int bz = blockIdx.z;
    A += (long)bz * sA;
    B += (long)bz * sB;
    const long coff = (long)bz * sC;

    const int m0 = blockIdx.y * 64;
    const int n0 = blockIdx.x * 128;
    const int tid = threadIdx.x;
    const int wave = tid >> 5, lane = tid & 31;
    const int wm = wave >> 2, wn = wave & 3;      // 2 x 4 wave grid
    const int r = lane & 15, hl = lane >> 4;

    // per-thread staging coordinates
    const int arow = tid >> 2, acol = (tid & 3) * 8;     // A: 64 x 32
    const int bkr = tid >> 4, bnc = (tid & 15) * 8;      // B: 32 x 128

    v8f acc00 = {}, acc01 = {}, acc10 = {}, acc11 = {};
    v8bf aR, bR0, bR1;

    // prologue: fetch K-step 0
    aR  = *(const v8bf*)(A + (long)(m0 + arow) * lda + acol);
    bR0 = *(const v8bf*)(B + (long)(bkr) * ldb + n0 + bnc);
    bR1 = *(const v8bf*)(B + (long)(16 + bkr) * ldb + n0 + bnc);

    for (int k0 = 32;; k0 += 32) {
        // commit staged registers to LDS
        *(v8bf*)&As[arow][acol] = aR;
#pragma unroll
        for (int i = 0; i < 8; ++i) Bt[bnc + i][bkr] = bR0[i];
#pragma unroll
        for (int i = 0; i < 8; ++i) Bt[bnc + i][16 + bkr] = bR1[i];
        __syncthreads();

        // prefetch next K-step while WMMAs consume LDS
        if (k0 < K) {
            aR  = *(const v8bf*)(A + (long)(m0 + arow) * lda + k0 + acol);
            bR0 = *(const v8bf*)(B + (long)(k0 + bkr) * ldb + n0 + bnc);
            bR1 = *(const v8bf*)(B + (long)(k0 + 16 + bkr) * ldb + n0 + bnc);
        }

        v16bf a0 = frag_a_ptr(&As[wm * 32 + r][0], 0);
        v16bf a1 = frag_a_ptr(&As[wm * 32 + 16 + r][0], 0);
        v16bf b0 = frag_b_ptr(&Bt[wn * 32][0], 40, 0);
        v16bf b1 = frag_b_ptr(&Bt[wn * 32 + 16][0], 40, 0);
        acc00 = wmma_bf16(a0, b0, acc00);
        acc01 = wmma_bf16(a0, b1, acc01);
        acc10 = wmma_bf16(a1, b0, acc10);
        acc11 = wmma_bf16(a1, b1, acc11);

        if (k0 >= K) break;
        __syncthreads();
    }

#pragma unroll
    for (int i = 0; i < 8; ++i) {
        long row0 = m0 + wm * 32 + i + 8 * hl;
        long row1 = row0 + 16;
        long c0 = n0 + wn * 32 + r;
        long c1 = c0 + 16;
        if (Cf) {
            Cf[coff + row0 * ldc + c0] = acc00[i];
            Cf[coff + row0 * ldc + c1] = acc01[i];
            Cf[coff + row1 * ldc + c0] = acc10[i];
            Cf[coff + row1 * ldc + c1] = acc11[i];
        }
        if (Cb) {
            Cb[coff + row0 * ldc + c0] = (bf16)acc00[i];
            Cb[coff + row0 * ldc + c1] = (bf16)acc01[i];
            Cb[coff + row1 * ldc + c0] = (bf16)acc10[i];
            Cb[coff + row1 * ldc + c1] = (bf16)acc11[i];
        }
    }
}

// ---------------------------------------------------------------------------
// Compression: per (b, block) softmax over M=128 tokens, weighted sum of
// c_proj, then layernorm -> k_comp (bf16) and v_comp (bf16).
// zc layout: [token][0:128]=c_proj, [128:256]=z.  128 threads = channels.
// ---------------------------------------------------------------------------
__global__ void __launch_bounds__(128)
compress_kernel(const float* __restrict__ zc, const float* __restrict__ b_comp,
                const float* __restrict__ gamma, const float* __restrict__ beta,
                bf16* __restrict__ k_bf, bf16* __restrict__ v_bf) {
    __shared__ float red[128], red2[128];
    const int t = threadIdx.x;
    const int blk = blockIdx.x & (N_COMP - 1);
    const int b = blockIdx.x >> 6;
    const float* base = zc + ((long)(b * S_SZ + blk * M_BLK)) * 256;

    float mx = -3.4e38f;
    for (int m = 0; m < M_BLK; ++m)
        mx = fmaxf(mx, base[m * 256 + 128 + t] + b_comp[m * 128 + t]);

    float Z = 0.f, acc = 0.f;
    for (int m = 0; m < M_BLK; ++m) {
        float e = __expf(base[m * 256 + 128 + t] + b_comp[m * 128 + t] - mx);
        Z += e;
        acc += e * base[m * 256 + t];
    }
    float comp = acc / Z;

    red[t] = comp; red2[t] = comp * comp;
    __syncthreads();
    for (int s2 = 64; s2 > 0; s2 >>= 1) {
        if (t < s2) { red[t] += red[t + s2]; red2[t] += red2[t + s2]; }
        __syncthreads();
    }
    float mu = red[0] * (1.f / 128.f);
    float var = red2[0] * (1.f / 128.f) - mu * mu;
    float kk = (comp - mu) * rsqrtf(var + 1e-6f) * gamma[t] + beta[t];

    long o = (long)blockIdx.x * 128 + t;
    k_bf[o] = (bf16)kk;
    v_bf[o] = (bf16)comp;
}

// ---------------------------------------------------------------------------
// Q layernorm + partial RoPE, in place on bf16 q (B,S,N_H,C).
// One wave32 per (b,s,h) row; lane owns 4 consecutive channels.
// ---------------------------------------------------------------------------
__global__ void __launch_bounds__(256)
qln_rope_kernel(bf16* __restrict__ q, const float* __restrict__ gamma,
                const float* __restrict__ beta) {
    const int wave = threadIdx.x >> 5, lane = threadIdx.x & 31;
    const long row = (long)blockIdx.x * 8 + wave;      // (b*S + s)*N_H + h
    const long s = (row >> 4) & (S_SZ - 1);
    bf16* p = q + row * C_DIM;

    float v[4];
#pragma unroll
    for (int i = 0; i < 4; ++i) v[i] = (float)p[lane * 4 + i];
    float sum = v[0] + v[1] + v[2] + v[3];
    float sq = v[0] * v[0] + v[1] * v[1] + v[2] * v[2] + v[3] * v[3];
#pragma unroll
    for (int m = 16; m >= 1; m >>= 1) {
        sum += __shfl_xor(sum, m, 32);
        sq += __shfl_xor(sq, m, 32);
    }
    float mu = sum * (1.f / 128.f);
    float var = sq * (1.f / 128.f) - mu * mu;
    float rs = rsqrtf(var + 1e-6f);
#pragma unroll
    for (int i = 0; i < 4; ++i) {
        int c = lane * 4 + i;
        v[i] = (v[i] - mu) * rs * gamma[c] + beta[c];
    }
    if (lane >= 16) {  // channels 64..127 get RoPE
#pragma unroll
        for (int j = 0; j < 2; ++j) {
            int c = lane * 4 + 2 * j;
            int i = (c - 64) >> 1;
            float freq = __powf(10000.f, -(float)(2 * i) * (1.f / 64.f));
            float sn, cs;
            __sincosf((float)s * freq, &sn, &cs);
            float e = v[2 * j], o = v[2 * j + 1];
            v[2 * j]     = e * cs - o * sn;
            v[2 * j + 1] = e * sn + o * cs;
        }
    }
#pragma unroll
    for (int i = 0; i < 4; ++i) p[lane * 4 + i] = (bf16)v[i];
}

// ---------------------------------------------------------------------------
// Attention over compressed KV.
// grid (S/128, 4, B); each block: 128 tokens x 4 heads. 8 waves, each wave
// owns 16 token rows. K tile staged via TDM (tensor_load_to_lds) with LDS
// row padding 128 -> 136 elements; V staged transposed for B-fragments.
// ---------------------------------------------------------------------------
#define KSTR 136
#define VSTR 72

__global__ void __launch_bounds__(256)
attn_kernel(const bf16* __restrict__ q, const bf16* __restrict__ k_bf,
            const bf16* __restrict__ v_bf, const float* __restrict__ sink_logits,
            bf16* __restrict__ out) {
    __shared__ bf16 kLds[N_COMP * KSTR];        // [block][channel], padded
    __shared__ bf16 vtLds[C_DIM * VSTR];        // [channel][block], padded
    __shared__ bf16 attnLds[8][16 * VSTR];      // per-wave 16x64 attn tile

    const int b = blockIdx.z;
    const int s0 = blockIdx.x * 128;
    const int tid = threadIdx.x, wave = tid >> 5, lane = tid & 31;
    const int r = lane & 15, hl = lane >> 4;

    // ---- K tile (64 x 128 bf16) via Tensor Data Mover, pad 4 DW per 64 DW row
    if (wave == 0) {
        unsigned long long ga =
            (unsigned long long)(size_t)(const void*)(k_bf + (long)b * N_COMP * C_DIM);
        unsigned int lds = (unsigned int)(size_t)(void*)kLds;
        u32x4_t g0;
        i32x8_t g1;
        i32x4_t gz4 = {0, 0, 0, 0};
        i32x8_t gz8 = {0, 0, 0, 0, 0, 0, 0, 0};
        g0[0] = 1u;                                       // count=1 (user D#)
        g0[1] = lds;                                      // lds_addr
        g0[2] = (unsigned int)(ga & 0xffffffffu);         // global_addr lo
        g0[3] = (unsigned int)((ga >> 32) & 0x1ffffffu) | (2u << 30);  // type=2
        // data_size=2B, pad_enable, pad_interval=64 DW, pad_amount=4 DW
        g1[0] = (int)((1u << 16) | (1u << 20) | (5u << 22) | (3u << 25));
        g1[1] = (int)(128u << 16);   // tensor_dim0 = 128
        g1[2] = (int)(64u << 16);    // tensor_dim1 = 64
        g1[3] = (int)(128u << 16);   // tile_dim0 = 128
        g1[4] = 64;                  // tile_dim1 = 64
        g1[5] = 128;                 // tensor_dim0_stride = 128
        g1[6] = 0;
        g1[7] = 0;
        __builtin_amdgcn_tensor_load_to_lds(g0, g1, gz4, gz4, gz8, 0);
    }
    // ---- V transposed into LDS (blocks -> K dim of second GEMM)
    for (int idx = tid; idx < N_COMP * C_DIM; idx += 256) {
        int blk = idx >> 7, c = idx & 127;
        vtLds[c * VSTR + blk] = v_bf[(long)b * N_COMP * C_DIM + idx];
    }
    if (wave == 0) __builtin_amdgcn_s_wait_tensorcnt(0);
    __syncthreads();

    const int row0 = s0 + wave * 16;
    const float scale = 0.08838834764831845f;   // 1/sqrt(128)

    for (int hi = 0; hi < 4; ++hi) {
        const int h = blockIdx.y * 4 + hi;
        const float sink = __expf(sink_logits[h]);

        // q A-fragments straight from global (row-major bf16, 16B aligned)
        const bf16* qrow =
            q + (((long)(b * S_SZ + row0 + r)) * N_H + h) * C_DIM;
        v16bf a0 = frag_a_ptr(qrow, 0);
        v16bf a1 = frag_a_ptr(qrow, 32);
        v16bf a2 = frag_a_ptr(qrow, 64);
        v16bf a3 = frag_a_ptr(qrow, 96);

        // scores: 16 tokens x 64 blocks, K = 128 channels
        v8f sc[4];
#pragma unroll
        for (int t = 0; t < 4; ++t) {
            v8f acc = {};
            const bf16* kb = kLds + (16 * t) * KSTR;
            acc = wmma_bf16(a0, frag_b_ptr(kb, KSTR, 0), acc);
            acc = wmma_bf16(a1, frag_b_ptr(kb, KSTR, 32), acc);
            acc = wmma_bf16(a2, frag_b_ptr(kb, KSTR, 64), acc);
            acc = wmma_bf16(a3, frag_b_ptr(kb, KSTR, 96), acc);
            sc[t] = acc;
        }

        // masked exp + sink normalization (reference: raw exp, no row max)
        float rowsum[8];
#pragma unroll
        for (int i = 0; i < 8; ++i) {
            int token = row0 + i + 8 * hl;
            float rsum = 0.f;
#pragma unroll
            for (int t = 0; t < 4; ++t) {
                int blk = 16 * t + r;
                float pv = (token >= blk * M_BLK) ? __expf(sc[t][i] * scale) : 0.f;
                sc[t][i] = pv;
                rsum += pv;
            }
            rowsum[i] = rsum;
        }
#pragma unroll
        for (int m = 1; m < 16; m <<= 1) {
#pragma unroll
            for (int i = 0; i < 8; ++i)
                rowsum[i] += __shfl_xor(rowsum[i], m, 32);
        }
#pragma unroll
        for (int i = 0; i < 8; ++i) {
            float inv = 1.f / (rowsum[i] + sink);
#pragma unroll
            for (int t = 0; t < 4; ++t)
                attnLds[wave][(i + 8 * hl) * VSTR + 16 * t + r] =
                    (bf16)(sc[t][i] * inv);
        }
        __syncthreads();

        // out = attn(16x64) @ v(64x128), K = 64 blocks
        v16bf aa0 = frag_a_ptr(&attnLds[wave][r * VSTR], 0);
        v16bf aa1 = frag_a_ptr(&attnLds[wave][r * VSTR], 32);
#pragma unroll
        for (int ct = 0; ct < 8; ++ct) {
            v8f acc = {};
            const bf16* vb = vtLds + (16 * ct) * VSTR;
            acc = wmma_bf16(aa0, frag_b_ptr(vb, VSTR, 0), acc);
            acc = wmma_bf16(aa1, frag_b_ptr(vb, VSTR, 32), acc);
#pragma unroll
            for (int i = 0; i < 8; ++i) {
                int token = row0 + i + 8 * hl;
                int c = 16 * ct + r;
                out[((long)(b * S_SZ + token)) * D_MODEL + h * C_DIM + c] =
                    (bf16)acc[i];
            }
        }
        __syncthreads();
    }
}

// ---------------------------------------------------------------------------
// Host launcher
// ---------------------------------------------------------------------------
extern "C" void kernel_launch(void* const* d_in, const int* in_sizes, int n_in,
                              void* d_out, int out_size, void* d_ws, size_t ws_size,
                              hipStream_t stream) {
    (void)in_sizes; (void)n_in; (void)out_size; (void)ws_size;

    const float* x    = (const float*)d_in[0];
    const float* wkv  = (const float*)d_in[1];
    const float* wz   = (const float*)d_in[2];
    const float* bcmp = (const float*)d_in[3];
    const float* wdq  = (const float*)d_in[4];
    const float* wuq  = (const float*)d_in[5];
    const float* odw  = (const float*)d_in[6];
    const float* ouw  = (const float*)d_in[7];
    const float* kvg  = (const float*)d_in[8];
    const float* kvb  = (const float*)d_in[9];
    const float* qg   = (const float*)d_in[10];
    const float* qb   = (const float*)d_in[11];
    const float* sink = (const float*)d_in[12];
    float* out = (float*)d_out;

    char* ws = (char*)d_ws;
    size_t off = 0;
    auto carve = [&](size_t bytes) -> void* {
        void* p = ws + off;
        off += (bytes + 255) & ~(size_t)255;
        return p;
    };

    bf16*  x_bf    = (bf16*)carve((size_t)ROWS * D_MODEL * 2);
    bf16*  wcat_bf = (bf16*)carve((size_t)D_MODEL * 256 * 2);
    bf16*  wdq_bf  = (bf16*)carve((size_t)D_MODEL * D_C * 2);
    bf16*  wuq_bf  = (bf16*)carve((size_t)D_C * (N_H * C_DIM) * 2);
    bf16*  odw_bf  = (bf16*)carve((size_t)N_GROUPS * D_G * D_G * 2);
    bf16*  ouw_bf  = (bf16*)carve((size_t)(N_GROUPS * D_G) * D_MODEL * 2);
    float* zc      = (float*)carve((size_t)ROWS * 256 * 4);
    bf16*  k_bf    = (bf16*)carve((size_t)B_SZ * N_COMP * C_DIM * 2);
    bf16*  v_bf    = (bf16*)carve((size_t)B_SZ * N_COMP * C_DIM * 2);
    bf16*  qlat    = (bf16*)carve((size_t)ROWS * D_C * 2);
    bf16*  qraw    = (bf16*)carve((size_t)ROWS * (N_H * C_DIM) * 2);
    bf16*  attn_o  = (bf16*)carve((size_t)ROWS * D_MODEL * 2);
    bf16*  down_bf = (bf16*)carve((size_t)ROWS * D_MODEL * 2);

    // 1. casts
    {
        long n = (long)ROWS * D_MODEL;
        cast_f32_bf16_kernel<<<(n + 255) / 256, 256, 0, stream>>>(x, x_bf, n);
        n = (long)D_MODEL * 256;
        build_wcat_kernel<<<(n + 255) / 256, 256, 0, stream>>>(wkv, wz, wcat_bf);
        n = (long)D_MODEL * D_C;
        cast_f32_bf16_kernel<<<(n + 255) / 256, 256, 0, stream>>>(wdq, wdq_bf, n);
        n = (long)D_C * N_H * C_DIM;
        cast_f32_bf16_kernel<<<(n + 255) / 256, 256, 0, stream>>>(wuq, wuq_bf, n);
        n = (long)N_GROUPS * D_G * D_G;
        cast_f32_bf16_kernel<<<(n + 255) / 256, 256, 0, stream>>>(odw, odw_bf, n);
        n = (long)(N_GROUPS * D_G) * D_MODEL;
        cast_f32_bf16_kernel<<<(n + 255) / 256, 256, 0, stream>>>(ouw, ouw_bf, n);
    }

    // 2. zc = x @ [w_kv | w_z]   (16384 x 2048) @ (2048 x 256) -> f32
    gemm_bf16_kernel<<<dim3(256 / 128, ROWS / 64, 1), 256, 0, stream>>>(
        x_bf, wcat_bf, zc, nullptr, D_MODEL, D_MODEL, 256, 256, 0, 0, 0);

    // 3. compression softmax + layernorm -> k_bf, v_bf
    compress_kernel<<<dim3(B_SZ * N_COMP), 128, 0, stream>>>(
        zc, bcmp, kvg, kvb, k_bf, v_bf);

    // 4. qlat = x @ w_dq  (-> bf16)
    gemm_bf16_kernel<<<dim3(D_C / 128, ROWS / 64, 1), 256, 0, stream>>>(
        x_bf, wdq_bf, nullptr, qlat, D_MODEL, D_MODEL, D_C, D_C, 0, 0, 0);

    // 5. qraw = qlat @ w_uq  (-> bf16, laid out (b,s,h,c))
    gemm_bf16_kernel<<<dim3((N_H * C_DIM) / 128, ROWS / 64, 1), 256, 0, stream>>>(
        qlat, wuq_bf, nullptr, qraw, D_C, D_C, N_H * C_DIM, N_H * C_DIM, 0, 0, 0);

    // 6. layernorm + partial rope, in place
    qln_rope_kernel<<<dim3((ROWS * N_H) / 8), 256, 0, stream>>>(qraw, qg, qb);

    // 7. attention -> attn_o bf16
    attn_kernel<<<dim3(S_SZ / 128, N_H / 4, B_SZ), 256, 0, stream>>>(
        qraw, k_bf, v_bf, sink, attn_o);

    // 8. grouped down-proj: 4 batched GEMMs (column-offset batch strides)
    gemm_bf16_kernel<<<dim3(D_G / 128, ROWS / 64, N_GROUPS), 256, 0, stream>>>(
        attn_o, odw_bf, nullptr, down_bf, D_G, D_MODEL, D_G, D_MODEL,
        (long)D_G, (long)D_G * D_G, (long)D_G);

    // 9. up-proj -> f32 output
    gemm_bf16_kernel<<<dim3(D_MODEL / 128, ROWS / 64, 1), 256, 0, stream>>>(
        down_bf, ouw_bf, out, nullptr, D_MODEL, D_MODEL, D_MODEL, D_MODEL,
        0, 0, 0);
}
